// PSAttentionBlock_88923002896878
// MI455X (gfx1250) — matrix operand order repaired and checked by reading, not compiled, not run on hardware
//
#include <hip/hip_runtime.h>
#include <hip/hip_bf16.h>

// ---------------------------------------------------------------------------
// PSAttentionBlock for MI455X (gfx1250, wave32, WMMA).
// B=16, C=KF=VF=128, H=W=128.  ~34 GFLOP vs ~270MB mandatory HBM traffic ->
// HBM-bound (~12us floor at 23.3 TB/s); all GEMMs run on
// v_wmma_f32_16x16x32_bf16, f32 accumulate/softmax.
// Pipeline (total ~738MB moved, ~32us):
//   1) proj:   Q,K = (folded qk_w) @ x   -> bf16 scratch (x staged/transposed
//              to LDS once per 128-pixel tile)
//   2) scores: per (b, 4-f chunk) S = Q_f K_f^T, causal exp-softmax,
//              accumulate over f in regs, deterministic partial store
//   3) reduce: s_sum = sum of 32 partials
//   4) vproj:  out = (v_w @ x + v_b) * s_sum   (v-projection fused into the
//              output kernel: saves the 268MB V round-trip vs storing V)
// ---------------------------------------------------------------------------

typedef __attribute__((ext_vector_type(16))) __bf16 v16bf;
typedef __attribute__((ext_vector_type(8)))  __bf16 v8bf;
typedef __attribute__((ext_vector_type(8)))  float  v8f;
typedef __attribute__((ext_vector_type(4)))  float  v4f;

#define HW 16384
#define LDS_STRIDE 136   // 128 + 8 bf16 pad: keeps 16B alignment, staggers banks

// A-fragment (16x32 bf16, M x K): lane L, elem e -> M=L%16, K=(e&8)*2+((L>=16)*8)+(e&7)
// src row-major [M][K], row stride `stride` (elements). 2x 16B loads.
__device__ __forceinline__ v16bf load_a_frag(const __bf16* __restrict__ src, int stride,
                                             int m_base, int k_base, int lane) {
  int m = lane & 15;
  int khalf = (lane >> 4) << 3;              // 0 or 8
  const __bf16* p = src + (size_t)(m_base + m) * stride + k_base + khalf;
  v8bf lo = *(const v8bf*)p;                 // K = k_base+khalf+0..7
  v8bf hi = *(const v8bf*)(p + 16);          // K = k_base+khalf+16..23
  v16bf r;
#pragma unroll
  for (int i = 0; i < 8; ++i) { r[i] = lo[i]; r[8 + i] = hi[i]; }
  return r;
}

// B-fragment (32x16 bf16, K x N) with B[k][n] = src[n][k] (contraction axis
// contiguous per n-row): lane L, elem e -> N=L%16, K=(e&8)*2+((L>=16)*8)+(e&7).
__device__ __forceinline__ v16bf load_b_frag(const __bf16* __restrict__ src, int stride,
                                             int n_base, int k_base, int lane) {
  int n = lane & 15;
  int khalf = (lane >> 4) << 3;              // 0 or 8
  const __bf16* p = src + (size_t)(n_base + n) * stride + k_base + khalf;
  v8bf lo = *(const v8bf*)p;
  v8bf hi = *(const v8bf*)(p + 16);
  v16bf r;
#pragma unroll
  for (int i = 0; i < 8; ++i) { r[i] = lo[i]; r[8 + i] = hi[i]; }
  return r;
}

// Stage + transpose one x tile: x[b][c][pix0+p] (f32, coalesced float4 reads)
// -> Xt[p][c] bf16 in LDS (B-fragment friendly: contraction axis c contiguous)
__device__ __forceinline__ void stage_x_tile(const float* __restrict__ xb, int pix0,
                                             __bf16* Xt, int tid) {
#pragma unroll 4
  for (int it = tid; it < 128 * 32; it += 256) {
    int c = it >> 5;
    int p4 = (it & 31) << 2;
    v4f vv = *(const v4f*)(xb + (size_t)c * HW + pix0 + p4);
#pragma unroll
    for (int j = 0; j < 4; ++j) Xt[(p4 + j) * LDS_STRIDE + c] = (__bf16)vv[j];
  }
}

// ---------------------------------------------------------------------------
// Kernel 1: q/k projections.  Out[f,pix] = sum_c W[f,c] x[b,c,pix].
// A = folded weights (qk_w[:, :C]+qk_w[:, C:], bf16, built in registers from
// global), B = transposed x tile in LDS.  8 waves x 16-row strips.
// ---------------------------------------------------------------------------
__global__ __launch_bounds__(256) void proj_kernel(
    const float* __restrict__ x, const float* __restrict__ qk_w,
    const float* __restrict__ qk_b, __bf16* __restrict__ Qbf,
    __bf16* __restrict__ Kbf) {
  __shared__ __bf16 Xt[128 * LDS_STRIDE];    // 34,816 B

  const int b = blockIdx.x >> 7;
  const int pix0 = (blockIdx.x & 127) << 7;
  const int tid = threadIdx.x;
  const int lane = tid & 31;
  const int wave = tid >> 5;

  stage_x_tile(x + (size_t)b * 128 * HW, pix0, Xt, tid);
  __syncthreads();

  const int fbase = wave << 4;
  const int m = lane & 15;
  const int khalf = (lane >> 4) << 3;
  v16bf aq[4], ak[4];
#pragma unroll
  for (int kc = 0; kc < 4; ++kc) {
    v16bf tq, tk;
#pragma unroll
    for (int e = 0; e < 16; ++e) {
      int c = (kc << 5) + ((e & 8) << 1) + khalf + (e & 7);
      size_t rq = (size_t)(fbase + m) * 256;
      size_t rk = (size_t)(128 + fbase + m) * 256;
      tq[e] = (__bf16)(qk_w[rq + c] + qk_w[rq + 128 + c]);
      tk[e] = (__bf16)(qk_w[rk + c] + qk_w[rk + 128 + c]);
    }
    aq[kc] = tq; ak[kc] = tk;
  }

  const int hi = lane >> 4;
  float bq[8], bk[8];
#pragma unroll
  for (int r = 0; r < 8; ++r) {
    int f = fbase + r + (hi << 3);
    bq[r] = qk_b[f];
    bk[r] = qk_b[128 + f];
  }

  for (int nt = 0; nt < 8; ++nt) {
    v8f accq = {0.f,0.f,0.f,0.f,0.f,0.f,0.f,0.f};
    v8f acck = {0.f,0.f,0.f,0.f,0.f,0.f,0.f,0.f};
#pragma unroll
    for (int kc = 0; kc < 4; ++kc) {
      v16bf bfr = load_b_frag(Xt, LDS_STRIDE, nt << 4, kc << 5, lane);
      accq = __builtin_amdgcn_wmma_f32_16x16x32_bf16(false, aq[kc], false, bfr,
                                                     (short)0, accq, false, false);
      acck = __builtin_amdgcn_wmma_f32_16x16x32_bf16(false, ak[kc], false, bfr,
                                                     (short)0, acck, false, false);
    }
    int pix = pix0 + (nt << 4) + (lane & 15);
#pragma unroll
    for (int r = 0; r < 8; ++r) {
      int f = fbase + r + (hi << 3);
      size_t o = ((size_t)(b * 128) + f) * HW + pix;
      Qbf[o] = (__bf16)(accq[r] + bq[r]);
      Kbf[o] = (__bf16)(acck[r] + bk[r]);
    }
  }
}

// ---------------------------------------------------------------------------
// Kernel 2: scores + causal exp-softmax, accumulated over a chunk of 4 f's.
// S = Q_f @ K_f^T (32 WMMAs per wave strip), scale 128^-0.5, tril exp,
// row-normalize via shfl_xor over the 16-lane half, accumulate in regs,
// one deterministic partial store (no fp atomics -> bit-stable replays).
// ---------------------------------------------------------------------------
__global__ __launch_bounds__(256) void scores_kernel(
    const __bf16* __restrict__ Qbf, const __bf16* __restrict__ Kbf,
    float* __restrict__ partial) {
  __shared__ __bf16 Kt[128 * LDS_STRIDE];    // 34,816 B

  const int b = blockIdx.x >> 5;
  const int fc = blockIdx.x & 31;
  const int tid = threadIdx.x;
  const int lane = tid & 31;
  const int wave = tid >> 5;
  const int hbase = wave << 4;
  const int hi = lane >> 4;
  const int n = lane & 15;

  v8f sacc[8];
#pragma unroll
  for (int gt = 0; gt < 8; ++gt) sacc[gt] = (v8f){0.f,0.f,0.f,0.f,0.f,0.f,0.f,0.f};

  for (int fi = 0; fi < 4; ++fi) {
    const int f = (fc << 2) + fi;
    const __bf16* qsrc = Qbf + ((size_t)(b * 128) + f) * HW;
    const __bf16* ksrc = Kbf + ((size_t)(b * 128) + f) * HW;

    __syncthreads();                         // protect Kt from previous iter
#pragma unroll 4
    for (int it = tid; it < 2048; it += 256) {
      int row = it >> 4;
      int col = (it & 15) << 3;
      *(v8bf*)(&Kt[row * LDS_STRIDE + col]) = *(const v8bf*)(ksrc + row * 128 + col);
    }
    // Overlap: pull next f's K panel toward L2/L0 while this f computes
    // (global_prefetch_b8; 128B granularity -> 512 lines of 64 elems)
    if (fi < 3) {
      const __bf16* knext = ksrc + HW;
      for (int it = tid; it < 512; it += 256)
        __builtin_prefetch(knext + (it << 6), 0, 1);
    }
    __syncthreads();

    v16bf afr[4];                            // Q A-frags straight from global
#pragma unroll
    for (int kc = 0; kc < 4; ++kc) afr[kc] = load_a_frag(qsrc, 128, hbase, kc << 5, lane);

    v8f ev[8];
#pragma unroll
    for (int gt = 0; gt < 8; ++gt) {
      v8f acc = {0.f,0.f,0.f,0.f,0.f,0.f,0.f,0.f};
#pragma unroll
      for (int kc = 0; kc < 4; ++kc) {
        v16bf bfr = load_b_frag(Kt, LDS_STRIDE, gt << 4, kc << 5, lane);
        acc = __builtin_amdgcn_wmma_f32_16x16x32_bf16(false, afr[kc], false, bfr,
                                                      (short)0, acc, false, false);
      }
#pragma unroll
      for (int r = 0; r < 8; ++r) {
        int h = hbase + r + (hi << 3);
        int g = (gt << 4) + n;
        float s = acc[r] * 0.088388347648318447f;  // 128^-0.5
        acc[r] = (h >= g) ? __expf(s) : 0.0f;      // exp * tril mask
      }
      ev[gt] = acc;
    }
#pragma unroll
    for (int r = 0; r < 8; ++r) {
      float rp = 0.f;
#pragma unroll
      for (int gt = 0; gt < 8; ++gt) rp += ev[gt][r];
      rp += __shfl_xor(rp, 1, 32);
      rp += __shfl_xor(rp, 2, 32);
      rp += __shfl_xor(rp, 4, 32);
      rp += __shfl_xor(rp, 8, 32);             // stays within the 16-lane half
      float inv = 1.0f / rp;
#pragma unroll
      for (int gt = 0; gt < 8; ++gt) sacc[gt][r] += ev[gt][r] * inv;
    }
  }

  float* dst = partial + ((size_t)(b * 32) + fc) * HW;
#pragma unroll
  for (int gt = 0; gt < 8; ++gt) {
#pragma unroll
    for (int r = 0; r < 8; ++r) {
      int h = hbase + r + (hi << 3);
      int g = (gt << 4) + n;
      dst[h * 128 + g] = sacc[gt][r];
    }
  }
}

// Kernel 3: s_sum[b,p] = sum of the 32 f-chunk partials (deterministic)
__global__ __launch_bounds__(256) void reduce_kernel(const float* __restrict__ partial,
                                                     float* __restrict__ s_sum) {
  int idx = blockIdx.x * 256 + threadIdx.x;  // 16 * 16384 total
  if (idx >= 16 * HW) return;
  int b = idx >> 14;
  int p = idx & (HW - 1);
  const float* src = partial + (size_t)b * 32 * HW + p;
  float s = 0.f;
#pragma unroll
  for (int j = 0; j < 32; ++j) s += src[j * HW];
  s_sum[idx] = s;
}

// ---------------------------------------------------------------------------
// Kernel 4 (fused): out[b,f,pix] = (sum_c v_w[f,c] x[b,c,pix] + v_b[f]) *
// s_sum[b,pix].  Same WMMA structure as proj; the s_sum multiplier depends
// only on pix -> one scalar load per lane per 16-pixel tile, broadcast over
// all 8 accumulator rows.  Saves the 268MB f32 V round-trip.
// ---------------------------------------------------------------------------
__global__ __launch_bounds__(256) void vproj_kernel(
    const float* __restrict__ x, const float* __restrict__ v_w,
    const float* __restrict__ v_b, const float* __restrict__ s_sum,
    float* __restrict__ out) {
  __shared__ __bf16 Xt[128 * LDS_STRIDE];

  const int b = blockIdx.x >> 7;
  const int pix0 = (blockIdx.x & 127) << 7;
  const int tid = threadIdx.x;
  const int lane = tid & 31;
  const int wave = tid >> 5;

  stage_x_tile(x + (size_t)b * 128 * HW, pix0, Xt, tid);
  __syncthreads();

  const int fbase = wave << 4;
  const int m = lane & 15;
  const int khalf = (lane >> 4) << 3;
  v16bf av[4];
#pragma unroll
  for (int kc = 0; kc < 4; ++kc) {
    v16bf tv;
#pragma unroll
    for (int e = 0; e < 16; ++e) {
      int c = (kc << 5) + ((e & 8) << 1) + khalf + (e & 7);
      tv[e] = (__bf16)(v_w[(size_t)(fbase + m) * 128 + c]);
    }
    av[kc] = tv;
  }

  const int hi = lane >> 4;
  float bv[8];
#pragma unroll
  for (int r = 0; r < 8; ++r) bv[r] = v_b[fbase + r + (hi << 3)];

  for (int nt = 0; nt < 8; ++nt) {
    v8f accv = {0.f,0.f,0.f,0.f,0.f,0.f,0.f,0.f};
#pragma unroll
    for (int kc = 0; kc < 4; ++kc) {
      v16bf bfr = load_b_frag(Xt, LDS_STRIDE, nt << 4, kc << 5, lane);
      accv = __builtin_amdgcn_wmma_f32_16x16x32_bf16(false, av[kc], false, bfr,
                                                     (short)0, accv, false, false);
    }
    int pix = pix0 + (nt << 4) + (lane & 15);
    float s = s_sum[((size_t)b << 14) + pix];  // broadcast over all rows
#pragma unroll
    for (int r = 0; r < 8; ++r) {
      int f = fbase + r + (hi << 3);
      out[((size_t)(b * 128) + f) * HW + pix] = (accv[r] + bv[r]) * s;
    }
  }
}

extern "C" void kernel_launch(void* const* d_in, const int* in_sizes, int n_in,
                              void* d_out, int out_size, void* d_ws, size_t ws_size,
                              hipStream_t stream) {
  const float* x    = (const float*)d_in[0];
  const float* qk_w = (const float*)d_in[1];
  const float* qk_b = (const float*)d_in[2];
  const float* v_w  = (const float*)d_in[3];
  const float* v_b  = (const float*)d_in[4];
  float* out = (float*)d_out;

  // Scratch layout (~161 MiB total)
  char* ws = (char*)d_ws;
  __bf16* Qbf   = (__bf16*)(ws);                               //  64 MiB
  __bf16* Kbf   = (__bf16*)(ws + (size_t)67108864);            //  64 MiB
  float*  part  = (float*)(ws + (size_t)134217728);            //  32 MiB
  float*  s_sum = (float*)(ws + (size_t)134217728 + 33554432); //   1 MiB

  proj_kernel<<<16 * 128, 256, 0, stream>>>(x, qk_w, qk_b, Qbf, Kbf);
  scores_kernel<<<16 * 32, 256, 0, stream>>>(Qbf, Kbf, part);
  reduce_kernel<<<1024, 256, 0, stream>>>(part, s_sum);
  vproj_kernel<<<16 * 128, 256, 0, stream>>>(x, v_w, v_b, s_sum, out);
}
